// FactorizedGenerator_24017457119342
// MI455X (gfx1250) — matrix operand order repaired and checked by reading, not compiled, not run on hardware
//
#include <hip/hip_runtime.h>

typedef __attribute__((ext_vector_type(16))) _Float16 v16h;
typedef __attribute__((ext_vector_type(8)))  _Float16 v8h;
typedef __attribute__((ext_vector_type(8)))  float    v8f;

#define DD   256   // data dim (independent MLPs)
#define LL   8     // latent dim (layer-1 K, zero-padded to 32)
#define HH   64    // hidden dim
#define W1P  40    // LDS pitch (f16) for W1^T rows: K padded 8->32, pitch 40 spreads banks
#define WP   72    // LDS pitch (f16) for W2^T/W3^T rows: K=64, pitch 72 spreads banks
#define AP   72    // activation scratch pitch (f16)
#define WAVES 8
#define RPB  128   // batch rows per workgroup (8 waves x 16 rows)

__device__ __forceinline__ v8f wmma16(v16h a, v16h b, v8f c) {
  // D = A(16x32 f16) * B(32x16 f16) + C(16x16 f32)
  return __builtin_amdgcn_wmma_f32_16x16x32_f16(false, a, false, b, (short)0, c, false, false);
}

__device__ __forceinline__ float silu(float x) {
  // x * sigmoid(x) with hardware v_rcp_f32 + v_exp_f32 (both TRANS ops,
  // co-execute with the matrix pipe) instead of IEEE division.
  return x * __builtin_amdgcn_rcpf(1.f + __expf(-x));
}

__global__ __launch_bounds__(256) void factorized_mlp_wmma(
    const float* __restrict__ z,  const float* __restrict__ W1, const float* __restrict__ b1,
    const float* __restrict__ W2, const float* __restrict__ b2,
    const float* __restrict__ W3, const float* __restrict__ b3,
    const float* __restrict__ Wo, const float* __restrict__ bo,
    float* __restrict__ out)
{
  __shared__ _Float16 sW1[HH * W1P];          // W1^T, K-contiguous, zero-padded K 8->32
  __shared__ _Float16 sW2[HH * WP];           // W2^T, K-contiguous
  __shared__ _Float16 sW3[HH * WP];           // W3^T, K-contiguous
  __shared__ _Float16 sWo[HH];
  __shared__ float    sB1[HH], sB2[HH], sB3[HH];
  __shared__ _Float16 sAct[WAVES][16 * AP];   // per-wave 16x64 activation tile

  const int tid  = threadIdx.x;
  const int d    = blockIdx.y;
  const int wave = tid >> 5;
  const int lane = tid & 31;
  const int m    = lane & 15;   // tile row (A) / tile col (B,C,D)
  const int hf   = lane >> 4;   // lane-half selector per WMMA VGPR layouts
  const int rowBase = blockIdx.x * RPB + wave * 16;

  // ---- stage weights into LDS as f16, transposed so K is contiguous per column ----
  for (int i = tid; i < HH * 32; i += 256) {
    int h = i >> 5, k = i & 31;
    sW1[h * W1P + k] = (k < LL) ? (_Float16)W1[(size_t)d * (LL * HH) + k * HH + h]
                                : (_Float16)0.f;
  }
  for (int i = tid; i < HH * HH; i += 256) {  // coalesced global read, strided LDS write
    int k = i >> 6, h = i & 63;
    sW2[h * WP + k] = (_Float16)W2[(size_t)d * (HH * HH) + i];
    sW3[h * WP + k] = (_Float16)W3[(size_t)d * (HH * HH) + i];
  }
  for (int i = tid; i < HH; i += 256) {
    sWo[i] = (_Float16)Wo[(size_t)d * HH + i];
    sB1[i] = b1[(size_t)d * HH + i];
    sB2[i] = b2[(size_t)d * HH + i];
    sB3[i] = b3[(size_t)d * HH + i];
  }
  __syncthreads();

  _Float16* act = sAct[wave];

  // ---- layer 1: A fragment (16x32, only K<8 nonzero) straight from z ----
  v16h a0 = {};
  if (hf == 0) {
    const float* zp = z + ((size_t)(rowBase + m) * DD + d) * LL;
    #pragma unroll
    for (int k = 0; k < LL; ++k) a0[k] = (_Float16)zp[k];
  }
  #pragma unroll
  for (int nt = 0; nt < 4; ++nt) {
    union { v16h v; v8h h[2]; } ub;
    const _Float16* wp = sW1 + (nt * 16 + m) * W1P + hf * 16;
    ub.h[0] = *(const v8h*)wp;
    ub.h[1] = *(const v8h*)(wp + 8);
    v8f acc = {};
    acc = wmma16(a0, ub.v, acc);
    float bias = sB1[nt * 16 + m];
    #pragma unroll
    for (int r = 0; r < 8; ++r)
      act[(r + hf * 8) * AP + nt * 16 + m] = (_Float16)silu(acc[r] + bias);
  }

  // ---- layers 2 & 3: 16x64 = sum over two 32-wide K chunks ----
  #pragma unroll
  for (int layer = 0; layer < 2; ++layer) {
    const _Float16* Wt = layer ? sW3 : sW2;
    const float*    Bb = layer ? sB3 : sB2;

    v16h aF[2];                               // read ALL of act before overwriting it
    #pragma unroll
    for (int kc = 0; kc < 2; ++kc) {
      union { v16h v; v8h h[2]; } ua;
      const _Float16* p = act + m * AP + kc * 32 + hf * 16;
      ua.h[0] = *(const v8h*)p;               // K rel {0..7} / {16..23}
      ua.h[1] = *(const v8h*)(p + 8);         // K rel {8..15} / {24..31}
      aF[kc] = ua.v;
    }

    v8f accs[4];
    #pragma unroll
    for (int nt = 0; nt < 4; ++nt) {
      v8f acc = {};
      #pragma unroll
      for (int kc = 0; kc < 2; ++kc) {
        union { v16h v; v8h h[2]; } ub;
        const _Float16* wp = Wt + (nt * 16 + m) * WP + kc * 32 + hf * 16;
        ub.h[0] = *(const v8h*)wp;
        ub.h[1] = *(const v8h*)(wp + 8);
        acc = wmma16(aF[kc], ub.v, acc);
      }
      accs[nt] = acc;
    }
    #pragma unroll
    for (int nt = 0; nt < 4; ++nt) {
      float bias = Bb[nt * 16 + m];
      #pragma unroll
      for (int r = 0; r < 8; ++r)
        act[(r + hf * 8) * AP + nt * 16 + m] = (_Float16)silu(accs[nt][r] + bias);
    }
  }

  // ---- output head (64 -> 1) as WMMA: B has Wo in column 0, zeros elsewhere ----
  {
    v16h aH[2];
    #pragma unroll
    for (int kc = 0; kc < 2; ++kc) {
      union { v16h v; v8h h[2]; } ua;
      const _Float16* p = act + m * AP + kc * 32 + hf * 16;
      ua.h[0] = *(const v8h*)p;
      ua.h[1] = *(const v8h*)(p + 8);
      aH[kc] = ua.v;
    }
    v8f acc = {};
    #pragma unroll
    for (int kc = 0; kc < 2; ++kc) {
      // B(32x16) column 0 = Wo[kc*32 .. kc*32+31]; lane n=0 holds K 0..15 (hf=0)
      // or K 16..31 (hf=1) of the chunk in the contiguous k order aH also uses.
      v16h bW = {};
      if (m == 0) {
        union { v16h v; v8h h[2]; } ub;
        const _Float16* wp = sWo + kc * 32 + hf * 16;
        ub.h[0] = *(const v8h*)wp;
        ub.h[1] = *(const v8h*)(wp + 8);
        bW = ub.v;
      }
      acc = wmma16(aH[kc], bW, acc);
    }
    // D fragment column 0 lives in lane 0 (rows 0..7) and lane 16 (rows 8..15).
    if (m == 0) {
      float bov = bo[d];
      #pragma unroll
      for (int r = 0; r < 8; ++r)
        out[(size_t)(rowBase + r + hf * 8) * DD + d] = acc[r] + bov;
    }
  }
}

extern "C" void kernel_launch(void* const* d_in, const int* in_sizes, int n_in,
                              void* d_out, int out_size, void* d_ws, size_t ws_size,
                              hipStream_t stream) {
  const float* z  = (const float*)d_in[0];
  const float* W1 = (const float*)d_in[1];
  const float* b1 = (const float*)d_in[2];
  const float* W2 = (const float*)d_in[3];
  const float* b2 = (const float*)d_in[4];
  const float* W3 = (const float*)d_in[5];
  const float* b3 = (const float*)d_in[6];
  const float* Wo = (const float*)d_in[7];
  const float* bo = (const float*)d_in[8];
  float* out = (float*)d_out;

  const int B = in_sizes[0] / (DD * LL);   // 4096
  dim3 grid(B / RPB, DD, 1);
  factorized_mlp_wmma<<<grid, dim3(256, 1, 1), 0, stream>>>(
      z, W1, b1, W2, b2, W3, b3, Wo, bo, out);
}
